// RenderMAEEncoder_1881195675791
// MI455X (gfx1250) — matrix operand order repaired and checked by reading, not compiled, not run on hardware
//
#include <hip/hip_runtime.h>
#include <hip/hip_bf16.h>

typedef __attribute__((ext_vector_type(16))) _Float16 v16h;
typedef __attribute__((ext_vector_type(8)))  float    v8f;

#define EMBED  384
#define HEADS  6
#define DH     64
#define TOK    128
#define BATCH  32
#define NPTS   2048
#define KNN    32
#define NTOKS  (BATCH * TOK)   // 4096

__device__ __forceinline__ float gelu_f(float x) {
  const float c = 0.7978845608028654f;
  float t = tanhf(c * (x + 0.044715f * x * x * x));
  return 0.5f * x * (1.0f + t);
}

// ---------------- WMMA 16x16x32 f16 fragment loaders ----------------
// A (16x32, MxK): lane&15 = row, lane>>4 selects K-half; VGPR j holds a K-pair.
__device__ __forceinline__ v16h frag_a_f32(const float* A, int lda, int row0, int k0, int lane) {
  int r = row0 + (lane & 15);
  int kh = lane >> 4;
  v16h a;
#pragma unroll
  for (int j = 0; j < 8; ++j) {
    int kk = k0 + ((j >> 2) << 4) + (kh << 3) + ((j & 3) << 1);
    const float* p = A + (size_t)r * lda + kk;
    a[2 * j]     = (_Float16)p[0];
    a[2 * j + 1] = (_Float16)p[1];
  }
  return a;
}

// A row-major f16: two contiguous 16B segments per fragment -> b128 loads
__device__ __forceinline__ v16h frag_a_f16(const _Float16* A, int lda, int row0, int k0, int lane) {
  int r = row0 + (lane & 15);
  int kh = lane >> 4;
  v16h a;
#pragma unroll
  for (int j = 0; j < 8; ++j) {
    int kk = k0 + ((j >> 2) << 4) + (kh << 3) + ((j & 3) << 1);
    const _Float16* p = A + (size_t)r * lda + kk;
    a[2 * j]     = p[0];
    a[2 * j + 1] = p[1];
  }
  return a;
}

// B (32x16, KxN), K-major f16 (strided): element B[k][n] at B[k*ldb+n]
__device__ __forceinline__ v16h frag_b_f16(const _Float16* B, int ldb, int k0, int col0, int lane) {
  int n = col0 + (lane & 15);
  int kh = lane >> 4;
  v16h b;
#pragma unroll
  for (int j = 0; j < 8; ++j) {
    int kk = k0 + ((j >> 2) << 4) + (kh << 3) + ((j & 3) << 1);
    b[2 * j]     = B[(size_t)kk * ldb + n];
    b[2 * j + 1] = B[(size_t)(kk + 1) * ldb + n];
  }
  return b;
}

// B stored transposed (n-major) f16: contiguous K-pairs -> b128 loads
__device__ __forceinline__ v16h frag_bt_f16(const _Float16* Bt, int ldb, int k0, int col0, int lane) {
  int n = col0 + (lane & 15);
  int kh = lane >> 4;
  v16h b;
#pragma unroll
  for (int j = 0; j < 8; ++j) {
    int kk = k0 + ((j >> 2) << 4) + (kh << 3) + ((j & 3) << 1);
    const _Float16* p = Bt + (size_t)n * ldb + kk;
    b[2 * j]     = p[0];
    b[2 * j + 1] = p[1];
  }
  return b;
}

#define WMMA_F16(a, b, c) \
  __builtin_amdgcn_wmma_f32_16x16x32_f16(false, (a), false, (b), (short)0, (c), false, false)

// ---------------- weight convert + transpose: Wt[n*K+k] = (f16)W[k*N+n] ----------------
__global__ void __launch_bounds__(256) cvt_t_kernel(const float* __restrict__ W,
                                                    _Float16* __restrict__ Wt,
                                                    int K, int N) {
  int i = blockIdx.x * 256 + threadIdx.x;
  if (i >= K * N) return;
  int k = i / N, n = i - k * N;
  Wt[(size_t)n * K + k] = (_Float16)W[i];
}

// ---------------- FPS ----------------
__global__ void __launch_bounds__(256) fps_kernel(const float* __restrict__ pts,
                                                  int* __restrict__ cidx,
                                                  float* __restrict__ centers) {
  int b = blockIdx.x, tid = threadIdx.x;
  __shared__ float dist[NPTS];
  __shared__ float rv[256];
  __shared__ int   ri[256];
  __shared__ float lastpt[3];
  const float* P = pts + (size_t)b * NPTS * 3;
  for (int i = tid; i < NPTS; i += 256) dist[i] = 3.4e38f;
  int last = 0;
  for (int it = 0; it < TOK; ++it) {
    if (tid == 0) {
      cidx[b * TOK + it] = last;
      float x = P[last * 3], y = P[last * 3 + 1], z = P[last * 3 + 2];
      lastpt[0] = x; lastpt[1] = y; lastpt[2] = z;
      centers[((size_t)b * TOK + it) * 3 + 0] = x;
      centers[((size_t)b * TOK + it) * 3 + 1] = y;
      centers[((size_t)b * TOK + it) * 3 + 2] = z;
    }
    __syncthreads();
    float bv = -1.0f; int bi = 0;
    for (int i = tid; i < NPTS; i += 256) {
      float dx = P[i * 3] - lastpt[0];
      float dy = P[i * 3 + 1] - lastpt[1];
      float dz = P[i * 3 + 2] - lastpt[2];
      float d  = dx * dx + dy * dy + dz * dz;
      float nd = fminf(dist[i], d);
      dist[i] = nd;
      if (nd > bv) { bv = nd; bi = i; }
    }
    rv[tid] = bv; ri[tid] = bi;
    __syncthreads();
    for (int s = 128; s > 0; s >>= 1) {
      if (tid < s) {
        if (rv[tid + s] > rv[tid] || (rv[tid + s] == rv[tid] && ri[tid + s] < ri[tid])) {
          rv[tid] = rv[tid + s]; ri[tid] = ri[tid + s];
        }
      }
      __syncthreads();
    }
    last = ri[0];
    __syncthreads();
  }
}

// ---------------- kNN grouping ----------------
__global__ void __launch_bounds__(256) knn_group_kernel(const float* __restrict__ pts,
                                                        const float* __restrict__ centers,
                                                        float* __restrict__ grouped) {
  int gid = blockIdx.x;           // b*TOK + m
  int b = gid >> 7;
  int tid = threadIdx.x;
  __shared__ float d[NPTS];
  __shared__ float rv[256];
  __shared__ int   ri[256];
  const float* P = pts + (size_t)b * NPTS * 3;
  float cx = centers[(size_t)gid * 3 + 0];
  float cy = centers[(size_t)gid * 3 + 1];
  float cz = centers[(size_t)gid * 3 + 2];
  for (int i = tid; i < NPTS; i += 256) {
    float dx = P[i * 3] - cx, dy = P[i * 3 + 1] - cy, dz = P[i * 3 + 2] - cz;
    d[i] = dx * dx + dy * dy + dz * dz;
  }
  __syncthreads();
  for (int k = 0; k < KNN; ++k) {
    float bv = 3.5e38f; int bi = 0;
    for (int i = tid; i < NPTS; i += 256) {
      if (d[i] < bv) { bv = d[i]; bi = i; }
    }
    rv[tid] = bv; ri[tid] = bi;
    __syncthreads();
    for (int s = 128; s > 0; s >>= 1) {
      if (tid < s) {
        if (rv[tid + s] < rv[tid] || (rv[tid + s] == rv[tid] && ri[tid + s] < ri[tid])) {
          rv[tid] = rv[tid + s]; ri[tid] = ri[tid + s];
        }
      }
      __syncthreads();
    }
    if (tid == 0) {
      int j = ri[0];
      d[j] = 3.4e38f;
      float* g = grouped + ((size_t)gid * KNN + k) * 3;
      g[0] = P[j * 3] - cx; g[1] = P[j * 3 + 1] - cy; g[2] = P[j * 3 + 2] - cz;
    }
    __syncthreads();
  }
}

// ---------------- fused group MLP encoder (3->128->256->384, maxpool) ----------------
__global__ void __launch_bounds__(256) encode_group_kernel(
    const float* __restrict__ grouped,
    const float* __restrict__ w1, const float* __restrict__ b1,
    const _Float16* __restrict__ w2t, const float* __restrict__ b2,  // [256][128] f16
    const _Float16* __restrict__ w3t, const float* __restrict__ b3,  // [384][256] f16
    float* __restrict__ out /* [NTOKS,384] */) {
  int gid = blockIdx.x;  // token
  int tid = threadIdx.x;
  int wave = tid >> 5, lane = tid & 31;
  __shared__ float    gpts[KNN * 3];
  __shared__ _Float16 h1[KNN * 128];   // 8KB
  __shared__ _Float16 h2[KNN * 256];   // 16KB
  __shared__ float    pool[2 * EMBED]; // 3KB

  if (tid < KNN * 3) gpts[tid] = grouped[(size_t)gid * KNN * 3 + tid];
  for (int i = tid; i < 2 * EMBED; i += 256) pool[i] = -3.4e38f;
  __syncthreads();

  // layer1 (K=3) scalar + ReLU
  for (int i = tid; i < KNN * 128; i += 256) {
    int p = i >> 7, c = i & 127;
    float v = b1[c] + gpts[p * 3 + 0] * w1[c] + gpts[p * 3 + 1] * w1[128 + c] +
              gpts[p * 3 + 2] * w1[256 + c];
    h1[i] = (_Float16)fmaxf(v, 0.0f);
  }
  __syncthreads();

  // layer2: [32,128] @ [128,256] via WMMA, ReLU
  for (int t = wave; t < 32; t += 8) {
    int mt = t & 1, nt = t >> 1;
    v8f acc = {};
    for (int k0 = 0; k0 < 128; k0 += 32) {
      v16h a  = frag_a_f16(h1, 128, mt * 16, k0, lane);
      v16h bb = frag_bt_f16(w2t, 128, k0, nt * 16, lane);
      acc = WMMA_F16(a, bb, acc);
    }
    int rr = (lane >> 4) << 3;
    int n = nt * 16 + (lane & 15);
#pragma unroll
    for (int r = 0; r < 8; ++r) {
      int m = mt * 16 + r + rr;
      h2[m * 256 + n] = (_Float16)fmaxf(acc[r] + b2[n], 0.0f);
    }
  }
  __syncthreads();

  // layer3: [32,256] @ [256,384] via WMMA + column max-pool over 32 points
  for (int t = wave; t < 48; t += 8) {
    int mt = t & 1, nt = t >> 1;
    v8f acc = {};
    for (int k0 = 0; k0 < 256; k0 += 32) {
      v16h a  = frag_a_f16(h2, 256, mt * 16, k0, lane);
      v16h bb = frag_bt_f16(w3t, 256, k0, nt * 16, lane);
      acc = WMMA_F16(a, bb, acc);
    }
    float mx = acc[0];
#pragma unroll
    for (int r = 1; r < 8; ++r) mx = fmaxf(mx, acc[r]);
    mx = fmaxf(mx, __shfl_xor(mx, 16));   // combine the two row halves of the tile
    if (lane < 16) pool[mt * EMBED + nt * 16 + lane] = mx;
  }
  __syncthreads();
  for (int c = tid; c < EMBED; c += 256)
    out[(size_t)gid * EMBED + c] = fmaxf(pool[c], pool[EMBED + c]) + b3[c];
}

// ---------------- positional embedding 3->128 GELU ->384 ----------------
__global__ void __launch_bounds__(128) pos_embed_kernel(
    const float* __restrict__ centers,
    const float* __restrict__ w1, const float* __restrict__ b1,
    const float* __restrict__ w2, const float* __restrict__ b2,
    float* __restrict__ pos) {
  int t = blockIdx.x, tid = threadIdx.x;
  __shared__ float hid[128];
  float c0 = centers[(size_t)t * 3 + 0];
  float c1 = centers[(size_t)t * 3 + 1];
  float c2 = centers[(size_t)t * 3 + 2];
  hid[tid] = gelu_f(c0 * w1[tid] + c1 * w1[128 + tid] + c2 * w1[256 + tid] + b1[tid]);
  __syncthreads();
  for (int c = tid; c < EMBED; c += 128) {
    float s = b2[c];
    for (int j = 0; j < 128; ++j) s += hid[j] * w2[(size_t)j * EMBED + c];
    pos[(size_t)t * EMBED + c] = s;
  }
}

// ---------------- (x [+ pos]) -> layernorm; templated output precision ----------------
template <typename TO>
__global__ void __launch_bounds__(128) add_ln_kernel(const float* __restrict__ x,
                                                     const float* __restrict__ pos,
                                                     const float* __restrict__ g,
                                                     const float* __restrict__ be,
                                                     TO* __restrict__ out) {
  int t = blockIdx.x, tid = threadIdx.x;
  __shared__ float red[128];
  const float* xp = x + (size_t)t * EMBED;
  float v[3];
#pragma unroll
  for (int i = 0; i < 3; ++i) {
    int c = tid + i * 128;
    v[i] = xp[c] + (pos ? pos[(size_t)t * EMBED + c] : 0.0f);
  }
  red[tid] = v[0] + v[1] + v[2];
  __syncthreads();
  for (int s = 64; s > 0; s >>= 1) { if (tid < s) red[tid] += red[tid + s]; __syncthreads(); }
  float mu = red[0] * (1.0f / EMBED);
  __syncthreads();
  float q = 0.0f;
#pragma unroll
  for (int i = 0; i < 3; ++i) { float d = v[i] - mu; q += d * d; }
  red[tid] = q;
  __syncthreads();
  for (int s = 64; s > 0; s >>= 1) { if (tid < s) red[tid] += red[tid + s]; __syncthreads(); }
  float inv = rsqrtf(red[0] * (1.0f / EMBED) + 1e-5f);
#pragma unroll
  for (int i = 0; i < 3; ++i) {
    int c = tid + i * 128;
    out[(size_t)t * EMBED + c] = (TO)((v[i] - mu) * inv * g[c] + be[c]);
  }
}

// ------- generic WMMA GEMM, 32x32 super-tile per wave, all-f16 operands -------
// C = act(A[M,K] @ B[K,N] + bias) (+resid);  A row-major f16, Bt [N][K] f16
template <int ACT, typename TC>  // ACT: 0 none, 1 relu, 2 gelu
__global__ void __launch_bounds__(256) gemm_kernel(const _Float16* __restrict__ A,
                                                   const _Float16* __restrict__ Bt,
                                                   const float* __restrict__ bias,
                                                   const float* __restrict__ resid,
                                                   TC* __restrict__ C,
                                                   int M, int N, int K) {
  int wave = threadIdx.x >> 5, lane = threadIdx.x & 31;
  int ntN = N >> 5;
  int tile = blockIdx.x * 8 + wave;
  if (tile >= (M >> 5) * ntN) return;
  int mt = tile / ntN, nt = tile % ntN;
  int row0 = mt << 5, col0 = nt << 5;
  v8f acc00 = {}, acc01 = {}, acc10 = {}, acc11 = {};
  for (int k0 = 0; k0 < K; k0 += 32) {
    v16h a0 = frag_a_f16(A, K, row0, k0, lane);
    v16h a1 = frag_a_f16(A, K, row0 + 16, k0, lane);
    v16h b0 = frag_bt_f16(Bt, K, k0, col0, lane);
    v16h b1 = frag_bt_f16(Bt, K, k0, col0 + 16, lane);
    acc00 = WMMA_F16(a0, b0, acc00);
    acc01 = WMMA_F16(a0, b1, acc01);
    acc10 = WMMA_F16(a1, b0, acc10);
    acc11 = WMMA_F16(a1, b1, acc11);
  }
  int rr = (lane >> 4) << 3;
  int nloc = lane & 15;
  auto epi = [&](v8f& acc, int rbase, int col) {
#pragma unroll
    for (int r = 0; r < 8; ++r) {
      int row = rbase + r + rr;
      float v = acc[r] + bias[col];
      if (ACT == 1) v = fmaxf(v, 0.0f);
      if (ACT == 2) v = gelu_f(v);
      if (resid) v += resid[(size_t)row * N + col];
      C[(size_t)row * N + col] = (TC)v;
    }
  };
  epi(acc00, row0,      col0 + nloc);
  epi(acc01, row0,      col0 + 16 + nloc);
  epi(acc10, row0 + 16, col0 + nloc);
  epi(acc11, row0 + 16, col0 + 16 + nloc);
}

// ---------------- fused attention per (batch, head), f16 qkv in / f16 out ----------------
__global__ void __launch_bounds__(256) attention_kernel(const _Float16* __restrict__ qkv,
                                                        _Float16* __restrict__ attn) {
  int bh = blockIdx.x;
  int b = bh / HEADS, h = bh % HEADS;
  int wave = threadIdx.x >> 5, lane = threadIdx.x & 31;
  __shared__ float S[TOK * TOK];  // 64KB scores/probs (kept fp32 for softmax)
  const _Float16* base = qkv + (size_t)b * TOK * (3 * EMBED);

  // phase1: S = (q @ k^T) * 0.125   (8x8 tiles, K=64)
  for (int t = wave; t < 64; t += 8) {
    int mt = t >> 3, nt = t & 7;
    v8f acc = {};
    for (int k0 = 0; k0 < DH; k0 += 32) {
      v16h a  = frag_a_f16(base + h * DH, 3 * EMBED, mt * 16, k0, lane);
      v16h bb = frag_bt_f16(base + EMBED + h * DH, 3 * EMBED, k0, nt * 16, lane);
      acc = WMMA_F16(a, bb, acc);
    }
    int rr = (lane >> 4) << 3;
    int n = nt * 16 + (lane & 15);
#pragma unroll
    for (int r = 0; r < 8; ++r) S[(mt * 16 + r + rr) * TOK + n] = acc[r] * 0.125f;
  }
  __syncthreads();

  // phase2: row softmax
  if (threadIdx.x < TOK) {
    float* row = S + threadIdx.x * TOK;
    float mx = -3.4e38f;
    for (int j = 0; j < TOK; ++j) mx = fmaxf(mx, row[j]);
    float sum = 0.0f;
    for (int j = 0; j < TOK; ++j) { float e = __expf(row[j] - mx); row[j] = e; sum += e; }
    float inv = 1.0f / sum;
    for (int j = 0; j < TOK; ++j) row[j] *= inv;
  }
  __syncthreads();

  // phase3: O = S @ v  (8x4 tiles, K=128)
  for (int t = wave; t < 32; t += 8) {
    int mt = t >> 2, nt = t & 3;
    v8f acc = {};
    for (int k0 = 0; k0 < TOK; k0 += 32) {
      v16h a  = frag_a_f32(S, TOK, mt * 16, k0, lane);
      v16h bb = frag_b_f16(base + 2 * EMBED + h * DH, 3 * EMBED, k0, nt * 16, lane);
      acc = WMMA_F16(a, bb, acc);
    }
    int rr = (lane >> 4) << 3;
    int col = h * DH + nt * 16 + (lane & 15);
#pragma unroll
    for (int r = 0; r < 8; ++r) {
      int row = mt * 16 + r + rr;
      attn[((size_t)b * TOK + row) * EMBED + col] = (_Float16)acc[r];
    }
  }
}

// ---------------- host ----------------
static inline int gemm_grid32(int M, int N) {
  int tiles = (M >> 5) * (N >> 5);
  return (tiles + 7) / 8;
}

extern "C" void kernel_launch(void* const* d_in, const int* in_sizes, int n_in,
                              void* d_out, int out_size, void* d_ws, size_t ws_size,
                              hipStream_t stream) {
  (void)in_sizes; (void)n_in; (void)out_size; (void)ws_size;
  const float* vis_pts = (const float*)d_in[0];
  const float* pe_w[3] = {(const float*)d_in[1], (const float*)d_in[2], (const float*)d_in[3]};
  const float* pe_b[3] = {(const float*)d_in[4], (const float*)d_in[5], (const float*)d_in[6]};
  const float* pos_w1 = (const float*)d_in[7];
  const float* pos_b1 = (const float*)d_in[8];
  const float* pos_w2 = (const float*)d_in[9];
  const float* pos_b2 = (const float*)d_in[10];
  const float* norm_g = (const float*)d_in[11 + 12 * 12 + 0];
  const float* norm_b = (const float*)d_in[11 + 12 * 12 + 1];

  // workspace carve-out (256B aligned)
  char* ws = (char*)d_ws;
  auto carve = [&ws](size_t bytes) {
    void* p = (void*)ws;
    ws += (bytes + 255) & ~(size_t)255;
    return p;
  };
  int*   cidx    = (int*)carve((size_t)NTOKS * sizeof(int));
  float* centers = (float*)carve((size_t)NTOKS * 3 * sizeof(float));
  float* grouped = (float*)carve((size_t)NTOKS * KNN * 3 * sizeof(float));
  float* x       = (float*)carve((size_t)NTOKS * EMBED * sizeof(float));
  float* pos     = (float*)carve((size_t)NTOKS * EMBED * sizeof(float));
  // f16 activation buffers (WMMA operands)
  _Float16* lnb16  = (_Float16*)carve((size_t)NTOKS * EMBED * sizeof(_Float16));
  _Float16* qkv16  = (_Float16*)carve((size_t)NTOKS * 3 * EMBED * sizeof(_Float16));
  _Float16* attn16 = (_Float16*)carve((size_t)NTOKS * EMBED * sizeof(_Float16));
  _Float16* hid16  = (_Float16*)carve((size_t)NTOKS * 4 * EMBED * sizeof(_Float16));
  // f16 transposed weights
  _Float16* pe2_16 = (_Float16*)carve((size_t)128 * 256 * sizeof(_Float16));
  _Float16* pe3_16 = (_Float16*)carve((size_t)256 * EMBED * sizeof(_Float16));
  _Float16 *wqkv16[12], *wo16[12], *w1_16[12], *w2_16[12];
  for (int i = 0; i < 12; ++i) {
    wqkv16[i] = (_Float16*)carve((size_t)EMBED * 3 * EMBED * sizeof(_Float16));
    wo16[i]   = (_Float16*)carve((size_t)EMBED * EMBED * sizeof(_Float16));
    w1_16[i]  = (_Float16*)carve((size_t)EMBED * 4 * EMBED * sizeof(_Float16));
    w2_16[i]  = (_Float16*)carve((size_t)4 * EMBED * EMBED * sizeof(_Float16));
  }

  auto cvt = [&](const float* W, _Float16* Wt, int K, int N) {
    int total = K * N;
    cvt_t_kernel<<<(total + 255) / 256, 256, 0, stream>>>(W, Wt, K, N);
  };
  cvt(pe_w[1], pe2_16, 128, 256);
  cvt(pe_w[2], pe3_16, 256, EMBED);
  for (int blk = 0; blk < 12; ++blk) {
    const float* const* B = (const float* const*)(d_in + 11 + blk * 12);
    cvt(B[2],  wqkv16[blk], EMBED, 3 * EMBED);
    cvt(B[4],  wo16[blk],   EMBED, EMBED);
    cvt(B[8],  w1_16[blk],  EMBED, 4 * EMBED);
    cvt(B[10], w2_16[blk],  4 * EMBED, EMBED);
  }

  fps_kernel<<<BATCH, 256, 0, stream>>>(vis_pts, cidx, centers);
  knn_group_kernel<<<NTOKS, 256, 0, stream>>>(vis_pts, centers, grouped);
  encode_group_kernel<<<NTOKS, 256, 0, stream>>>(grouped, pe_w[0], pe_b[0], pe2_16, pe_b[1],
                                                 pe3_16, pe_b[2], x);
  pos_embed_kernel<<<NTOKS, 128, 0, stream>>>(centers, pos_w1, pos_b1, pos_w2, pos_b2, pos);

  for (int blk = 0; blk < 12; ++blk) {
    const float* const* B = (const float* const*)(d_in + 11 + blk * 12);
    const float *g1 = B[0], *be1 = B[1], *bqkv = B[3], *bo = B[5];
    const float *g2 = B[6], *be2 = B[7], *b1 = B[9], *b2 = B[11];

    // lnb16 = ln1(x + pos)  (f16 for WMMA-A)
    add_ln_kernel<_Float16><<<NTOKS, 128, 0, stream>>>(x, pos, g1, be1, lnb16);
    // qkv16 = lnb16 @ Wqkv + b  (f16 out)
    gemm_kernel<0, _Float16><<<gemm_grid32(NTOKS, 3 * EMBED), 256, 0, stream>>>(
        lnb16, wqkv16[blk], bqkv, nullptr, qkv16, NTOKS, 3 * EMBED, EMBED);
    // attn16 = softmax(q k^T / 8) v  (f16 out)
    attention_kernel<<<BATCH * HEADS, 256, 0, stream>>>(qkv16, attn16);
    // x = x + attn16 @ Wo + bo  (fp32 residual accumulation)
    gemm_kernel<0, float><<<gemm_grid32(NTOKS, EMBED), 256, 0, stream>>>(
        attn16, wo16[blk], bo, x, x, NTOKS, EMBED, EMBED);
    // lnb16 = ln2(x)
    add_ln_kernel<_Float16><<<NTOKS, 128, 0, stream>>>(x, nullptr, g2, be2, lnb16);
    // hid16 = gelu(lnb16 @ W1 + b1)  (f16 out)
    gemm_kernel<2, _Float16><<<gemm_grid32(NTOKS, 4 * EMBED), 256, 0, stream>>>(
        lnb16, w1_16[blk], b1, nullptr, hid16, NTOKS, 4 * EMBED, EMBED);
    // x = x + hid16 @ W2 + b2  (fp32 residual accumulation)
    gemm_kernel<0, float><<<gemm_grid32(NTOKS, EMBED), 256, 0, stream>>>(
        hid16, w2_16[blk], b2, x, x, NTOKS, EMBED, 4 * EMBED);
  }

  add_ln_kernel<float><<<NTOKS, 128, 0, stream>>>(x, nullptr, norm_g, norm_b, (float*)d_out);
}
